// StatefulCausalDiffAttention_12558484374040
// MI455X (gfx1250) — matrix operand order repaired and checked by reading, not compiled, not run on hardware
//
#include <hip/hip_runtime.h>

// ---------------- problem constants ----------------
#define H_      16
#define T_      2048
#define DIM_    2048
#define DK_     64
#define DE_     128           // 2*DK = 2*DV
#define STATE_  8
#define LAMBDA_INIT_ 0.6192834728526787f   // 0.8 - 0.6*exp(-1.2)
#define OUT_SCALE_   0.3807165271473213f   // 1 - LAMBDA_INIT
#define EPS_    1e-5f

// ---------------- vector types ----------------
typedef __bf16 v8bf  __attribute__((ext_vector_type(8)));
typedef __bf16 v16bf __attribute__((ext_vector_type(16)));
typedef float  v8f   __attribute__((ext_vector_type(8)));
typedef int    gvi4  __attribute__((vector_size(16)));   // matches builtin param

#define WMMA_BF16(a, b, c) \
  __builtin_amdgcn_wmma_f32_16x16x32_bf16(false, (a), false, (b), (short)0, (c), false, false)

// CDNA5 async global->LDS copy (ASYNCcnt path), guarded so a missing builtin
// degrades to a plain vector copy instead of breaking the build.
#if defined(__has_builtin)
#if __has_builtin(__builtin_amdgcn_global_load_async_to_lds_b128)
#define HAVE_ASYNC_LDS 1
#endif
#endif

// ------------- WMMA operand loaders (ISA 7.12.2 layouts, wave32) -------------
// A (16x32, bf16): lane L holds row M=L&15; lanes<16: K in {0..7,16..23},
// lanes>=16: +8. Two 16B loads per lane from a row-major [16][rowStride] source.
__device__ __forceinline__ v16bf load_A(const __bf16* base, int rowStride) {
  int lane = threadIdx.x & 31, lo = lane & 15, hi = lane >> 4;
  const __bf16* p = base + lo * rowStride;
  v8bf x0 = *(const v8bf*)(p + hi * 8);
  v8bf x1 = *(const v8bf*)(p + 16 + hi * 8);
  v16bf r;
#pragma unroll
  for (int i = 0; i < 8; ++i) { r[i] = x0[i]; r[8 + i] = x1[i]; }
  return r;
}
// B (32x16 KxN, bf16): lane L holds col N=L&15; lanes<16: K in {0..7,16..23},
// lanes>=16: +8 (same K order as A). Source stored row=N, col=K.
__device__ __forceinline__ v16bf load_B(const __bf16* base, int rowStride) {
  int lane = threadIdx.x & 31, lo = lane & 15, hi = lane >> 4;
  const __bf16* p = base + (size_t)lo * rowStride + hi * 8;
  v8bf x0 = *(const v8bf*)(p);
  v8bf x1 = *(const v8bf*)(p + 16);
  v16bf r;
#pragma unroll
  for (int i = 0; i < 8; ++i) { r[i] = x0[i]; r[8 + i] = x1[i]; }
  return r;
}

// ---------------- reductions (wave32; C-layout rows live in 16-lane halves) ---
__device__ __forceinline__ float halfMax(float v) {
  v = fmaxf(v, __shfl_xor(v, 1, 32));
  v = fmaxf(v, __shfl_xor(v, 2, 32));
  v = fmaxf(v, __shfl_xor(v, 4, 32));
  v = fmaxf(v, __shfl_xor(v, 8, 32));
  return v;
}
__device__ __forceinline__ float halfSum(float v) {
  v += __shfl_xor(v, 1, 32);
  v += __shfl_xor(v, 2, 32);
  v += __shfl_xor(v, 4, 32);
  v += __shfl_xor(v, 8, 32);
  return v;
}
__device__ __forceinline__ float fullSum(float v) {
  v = halfSum(v);
  v += __shfl_xor(v, 16, 32);
  return v;
}

// ======================= kernel 0: x f32 -> bf16 (one-time convert) ==========
__global__ __launch_bounds__(256) void xcvt_kernel(const float* __restrict__ x,
                                                   __bf16* __restrict__ Xb) {
  size_t i = ((size_t)blockIdx.x * 256 + threadIdx.x) * 8;
  float4 f0 = *(const float4*)(x + i);
  float4 f1 = *(const float4*)(x + i + 4);
  v8bf h;
  h[0] = (__bf16)f0.x; h[1] = (__bf16)f0.y; h[2] = (__bf16)f0.z; h[3] = (__bf16)f0.w;
  h[4] = (__bf16)f1.x; h[5] = (__bf16)f1.y; h[6] = (__bf16)f1.z; h[7] = (__bf16)f1.w;
  *(v8bf*)(Xb + i) = h;
}

// ======================= kernel 1: weight transpose f32 -> bf16 ==============
// W[h][k][n] (k=2048, n=128) -> Wt[h][n][k]; 6 tensors, order Wq,Wk,Wv,Wq_s,Wk_s,Wv_s
__global__ __launch_bounds__(256) void wt_transpose_kernel(
    const float* W0, const float* W1, const float* W2,
    const float* W3, const float* W4, const float* W5, __bf16* out) {
  __shared__ __align__(16) float tile[32][33];
  int z = blockIdx.z, w = z >> 4, h = z & 15;
  const float* W;
  switch (w) {
    case 0: W = W0; break; case 1: W = W1; break; case 2: W = W2; break;
    case 3: W = W3; break; case 4: W = W4; break; default: W = W5; break;
  }
  const float* src = W + (size_t)h * DIM_ * DE_;
  __bf16* dst = out + ((size_t)w * H_ + h) * ((size_t)DE_ * DIM_);
  int tx = threadIdx.x, ty = threadIdx.y;
  int n0 = blockIdx.x * 32, k0 = blockIdx.y * 32;
#pragma unroll
  for (int i = 0; i < 4; ++i) {
    int k = k0 + ty + i * 8;
    tile[ty + i * 8][tx] = src[(size_t)k * DE_ + n0 + tx];
  }
  __syncthreads();
#pragma unroll
  for (int i = 0; i < 4; ++i) {
    int n = n0 + ty + i * 8;
    dst[(size_t)n * DIM_ + k0 + tx] = (__bf16)tile[tx][ty + i * 8];
  }
}

// ======================= kernel 2: QKV projection GEMM (bf16 WMMA) ===========
// grid: (row-tile 0..127, head 0..15, type 0..2). 256 threads = 8 waves; each
// wave owns one 16-col N-tile. Q/K stored natural [h][t][128]; V transposed
// [h][n][2048]. Tiles 0 and 127 additionally accumulate with the *_s weights.
// A-tile staged from pre-converted bf16 x via async global->LDS when available.
__global__ __launch_bounds__(256) void proj_kernel(
    const __bf16* __restrict__ Xb, const __bf16* __restrict__ wt,
    __bf16* __restrict__ Qb, __bf16* __restrict__ Kb, __bf16* __restrict__ Vt) {
  __shared__ __align__(16) __bf16 ldsA[16 * 128];
  int rt = blockIdx.x, h = blockIdx.y, type = blockIdx.z;
  int tid = threadIdx.x, wv = tid >> 5, lane = tid & 31, lo = lane & 15, hi = lane >> 4;
  const size_t WT_TENSOR = (size_t)H_ * DE_ * DIM_;
  const __bf16* Wn  = wt + (size_t)type       * WT_TENSOR + ((size_t)h * DE_ + wv * 16) * DIM_;
  const __bf16* Wsn = wt + (size_t)(type + 3) * WT_TENSOR + ((size_t)h * DE_ + wv * 16) * DIM_;
  bool straddle = (rt == 0) || (rt == 127);

  v8f acc  = (v8f)0.0f;
  v8f accS = (v8f)0.0f;

  int srow = tid >> 4;            // staging: 16 rows x 16 col-octets (16B each)
  int scp  = (tid & 15) * 8;
  const __bf16* xrow = Xb + (size_t)(rt * 16 + srow) * DIM_ + scp;
  __bf16* ldst = ldsA + srow * 128 + scp;

  auto stage = [&](int kb) {
#ifdef HAVE_ASYNC_LDS
    __builtin_amdgcn_global_load_async_to_lds_b128(
        (gvi4*)(xrow + kb), (gvi4*)ldst, 0, 0);
    asm volatile("s_wait_asynccnt 0x0" ::: "memory");
#else
    *(v8bf*)ldst = *(const v8bf*)(xrow + kb);
#endif
  };

  if (!straddle) {
    for (int kb = 0; kb < DIM_; kb += 128) {
      stage(kb);
      __syncthreads();
#pragma unroll
      for (int s = 0; s < 128; s += 32) {
        v16bf a = load_A(ldsA + s, 128);
        v16bf b = load_B(Wn + kb + s, DIM_);
        acc = WMMA_BF16(a, b, acc);
      }
      __syncthreads();
    }
  } else {
    for (int kb = 0; kb < DIM_; kb += 128) {
      stage(kb);
      __syncthreads();
#pragma unroll
      for (int s = 0; s < 128; s += 32) {
        v16bf a = load_A(ldsA + s, 128);
        v16bf b = load_B(Wn + kb + s, DIM_);
        acc = WMMA_BF16(a, b, acc);
        v16bf bs = load_B(Wsn + kb + s, DIM_);
        accS = WMMA_BF16(a, bs, accS);
      }
      __syncthreads();
    }
  }

  int n = wv * 16 + lo;
  if (type == 0) {
#pragma unroll
    for (int r = 0; r < 8; ++r) {
      int t = rt * 16 + r + 8 * hi;
      bool isState = (t < STATE_) || (t >= T_ - STATE_);
      Qb[((size_t)h * T_ + t) * DE_ + n] = (__bf16)(isState ? accS[r] : acc[r]);
    }
  } else if (type == 1) {
#pragma unroll
    for (int r = 0; r < 8; ++r) {
      int t = rt * 16 + r + 8 * hi;
      bool isState = (t < STATE_) || (t >= T_ - STATE_);
      Kb[((size_t)h * T_ + t) * DE_ + n] = (__bf16)(isState ? accS[r] : acc[r]);
    }
  } else {
#pragma unroll
    for (int r = 0; r < 8; ++r) {
      int t = rt * 16 + r + 8 * hi;
      bool isState = (t < STATE_) || (t >= T_ - STATE_);
      Vt[((size_t)h * DE_ + n) * T_ + t] = (__bf16)(isState ? accS[r] : acc[r]);
    }
  }
}

// ======================= kernel 3: differential flash attention ==============
// 128 threads = 4 waves; each wave handles one (head, 16-query block).
__global__ __launch_bounds__(128) void attn_kernel(
    const __bf16* __restrict__ Qb, const __bf16* __restrict__ Kb,
    const __bf16* __restrict__ Vt,
    const float* __restrict__ lq1, const float* __restrict__ lq2,
    const float* __restrict__ lk1, const float* __restrict__ lk2,
    const float* __restrict__ gamma, const float* __restrict__ beta,
    float* __restrict__ out) {
  __shared__ __align__(16) __bf16 Pl[4][2][16 * 32];
  int tid = threadIdx.x, wv = tid >> 5, lane = tid & 31, lo = lane & 15, hi = lane >> 4;
  int task = blockIdx.x * 4 + wv;
  int h = task >> 7, qb = task & 127;

  // lambda = exp(lq1.lk1) - exp(lq2.lk2) + lambda_init   (full-wave reduction)
  float d1 = lq1[h * DK_ + lane] * lk1[h * DK_ + lane]
           + lq1[h * DK_ + lane + 32] * lk1[h * DK_ + lane + 32];
  float d2 = lq2[h * DK_ + lane] * lk2[h * DK_ + lane]
           + lq2[h * DK_ + lane + 32] * lk2[h * DK_ + lane + 32];
  float lam = __expf(fullSum(d1)) - __expf(fullSum(d2)) + LAMBDA_INIT_;

  // Q operands: q1 = d[0:64), q2 = d[64:128)
  const __bf16* qrow = Qb + ((size_t)h * T_ + qb * 16) * DE_;
  v16bf q1k0 = load_A(qrow + 0,  DE_);
  v16bf q1k1 = load_A(qrow + 32, DE_);
  v16bf q2k0 = load_A(qrow + 64, DE_);
  v16bf q2k1 = load_A(qrow + 96, DE_);

  v8f O1[8], O2[8];
#pragma unroll
  for (int nt = 0; nt < 8; ++nt) { O1[nt] = (v8f)0.0f; O2[nt] = (v8f)0.0f; }
  float m1[8], l1[8], m2[8], l2[8];
#pragma unroll
  for (int r = 0; r < 8; ++r) { m1[r] = -1e30f; l1[r] = 0.f; m2[r] = -1e30f; l2[r] = 0.f; }

  __bf16* P1 = &Pl[wv][0][0];
  __bf16* P2 = &Pl[wv][1][0];
  const __bf16* vbase = Vt + (size_t)h * DE_ * T_;
  int kbmax = (qb * 16 + 15) >> 5;

  for (int kb = 0; kb <= kbmax; ++kb) {
    int key0 = kb * 32;
    const __bf16* krow = Kb + ((size_t)h * T_ + key0) * DE_;
    if (kb < kbmax) {   // prefetch next K/V tiles (global_prefetch_b8)
      __builtin_prefetch(krow + 32 * DE_, 0, 0);
      __builtin_prefetch(vbase + key0 + 32, 0, 0);
    }
    // ---- S tiles: s?a = keys[0:16), s?b = keys[16:32) -----------------------
    v8f s1a = (v8f)0.0f, s1b = (v8f)0.0f, s2a = (v8f)0.0f, s2b = (v8f)0.0f;
    {
      v16bf b0 = load_B(krow + 0,  DE_);
      v16bf b1 = load_B(krow + 32, DE_);
      s1a = WMMA_BF16(q1k0, b0, s1a); s1a = WMMA_BF16(q1k1, b1, s1a);
      v16bf b2 = load_B(krow + 64, DE_);
      v16bf b3 = load_B(krow + 96, DE_);
      s2a = WMMA_BF16(q2k0, b2, s2a); s2a = WMMA_BF16(q2k1, b3, s2a);
      const __bf16* krow2 = krow + 16 * DE_;
      b0 = load_B(krow2 + 0,  DE_);  b1 = load_B(krow2 + 32, DE_);
      s1b = WMMA_BF16(q1k0, b0, s1b); s1b = WMMA_BF16(q1k1, b1, s1b);
      b2 = load_B(krow2 + 64, DE_);  b3 = load_B(krow2 + 96, DE_);
      s2b = WMMA_BF16(q2k0, b2, s2b); s2b = WMMA_BF16(q2k1, b3, s2b);
    }
    // ---- scale + causal mask ------------------------------------------------
    bool needMask = (kb == kbmax);
#pragma unroll
    for (int r = 0; r < 8; ++r) {
      int t = qb * 16 + r + 8 * hi;
      float e0 = s1a[r] * 0.125f, e1 = s1b[r] * 0.125f;
      float e2 = s2a[r] * 0.125f, e3 = s2b[r] * 0.125f;
      if (needMask) {
        if (key0 + lo > t)       { e0 = -1e30f; e2 = -1e30f; }
        if (key0 + 16 + lo > t)  { e1 = -1e30f; e3 = -1e30f; }
      }
      s1a[r] = e0; s1b[r] = e1; s2a[r] = e2; s2b[r] = e3;
    }
    // ---- online softmax (both attentions); P -> LDS as bf16 A-layout src ----
    float al1[8], al2[8];
#pragma unroll
    for (int r = 0; r < 8; ++r) {
      int prow = (r + 8 * hi) * 32;
      {
        float mn = fmaxf(m1[r], halfMax(fmaxf(s1a[r], s1b[r])));
        float alpha = __expf(m1[r] - mn);
        float pa = __expf(s1a[r] - mn), pb = __expf(s1b[r] - mn);
        P1[prow + lo] = (__bf16)pa;
        P1[prow + 16 + lo] = (__bf16)pb;
        l1[r] = l1[r] * alpha + halfSum(pa + pb);
        m1[r] = mn; al1[r] = alpha;
      }
      {
        float mn = fmaxf(m2[r], halfMax(fmaxf(s2a[r], s2b[r])));
        float alpha = __expf(m2[r] - mn);
        float pa = __expf(s2a[r] - mn), pb = __expf(s2b[r] - mn);
        P2[prow + lo] = (__bf16)pa;
        P2[prow + 16 + lo] = (__bf16)pb;
        l2[r] = l2[r] * alpha + halfSum(pa + pb);
        m2[r] = mn; al2[r] = alpha;
      }
    }
#pragma unroll
    for (int nt = 0; nt < 8; ++nt)
#pragma unroll
      for (int r = 0; r < 8; ++r) { O1[nt][r] *= al1[r]; O2[nt][r] *= al2[r]; }

    asm volatile("" ::: "memory");       // order P stores before A-layout loads
    v16bf pA1 = load_A(P1, 32);
    v16bf pA2 = load_A(P2, 32);
    asm volatile("" ::: "memory");

    // ---- O += P * V  (V transposed: row = n, col = key) ---------------------
    const __bf16* vrow = vbase + key0;
#pragma unroll
    for (int nt = 0; nt < 8; ++nt) {
      v16bf vB = load_B(vrow + (size_t)nt * 16 * T_, T_);
      O1[nt] = WMMA_BF16(pA1, vB, O1[nt]);
      O2[nt] = WMMA_BF16(pA2, vB, O2[nt]);
    }
  }

  // ---- epilogue: differential combine + layernorm + gamma/beta --------------
  float g[8], bt[8];
#pragma unroll
  for (int nt = 0; nt < 8; ++nt) {
    g[nt]  = gamma[h * DE_ + nt * 16 + lo];
    bt[nt] = beta[h * DE_ + nt * 16 + lo];
  }
#pragma unroll
  for (int r = 0; r < 8; ++r) {
    float i1 = 1.0f / l1[r], i2 = 1.0f / l2[r];
    float sum = 0.0f;
#pragma unroll
    for (int nt = 0; nt < 8; ++nt) {
      float a = O1[nt][r] * i1 - lam * (O2[nt][r] * i2);
      O1[nt][r] = a;
      sum += a;
    }
    float mu = halfSum(sum) * (1.0f / 128.0f);
    float vs = 0.0f;
#pragma unroll
    for (int nt = 0; nt < 8; ++nt) { float d = O1[nt][r] - mu; vs += d * d; }
    float inv = rsqrtf(halfSum(vs) * (1.0f / 128.0f) + EPS_);
    int t = qb * 16 + r + 8 * hi;
    float* op = out + ((size_t)h * T_ + t) * DE_;
#pragma unroll
    for (int nt = 0; nt < 8; ++nt)
      op[nt * 16 + lo] = ((O1[nt][r] - mu) * inv * g[nt] + bt[nt]) * OUT_SCALE_;
  }
}

// =============================== host launcher ===============================
extern "C" void kernel_launch(void* const* d_in, const int* in_sizes, int n_in,
                              void* d_out, int out_size, void* d_ws, size_t ws_size,
                              hipStream_t stream) {
  (void)in_sizes; (void)n_in; (void)out_size;
  const float* x    = (const float*)d_in[0];
  const float* Wq   = (const float*)d_in[1];
  const float* Wk   = (const float*)d_in[2];
  const float* Wv   = (const float*)d_in[3];
  const float* Wqs  = (const float*)d_in[4];
  const float* Wks  = (const float*)d_in[5];
  const float* Wvs  = (const float*)d_in[6];
  const float* lq1  = (const float*)d_in[7];
  const float* lq2  = (const float*)d_in[8];
  const float* lk1  = (const float*)d_in[9];
  const float* lk2  = (const float*)d_in[10];
  const float* gam  = (const float*)d_in[11];
  const float* bet  = (const float*)d_in[12];

  const size_t WT_ELE = 6ull * H_ * DE_ * DIM_;   // transposed bf16 weights
  const size_t X_ELE  = (size_t)T_ * DIM_;        // bf16 copy of x
  const size_t QK_ELE = (size_t)H_ * T_ * DE_;    // Q / K / Vt bf16 buffers
  if (ws_size < (WT_ELE + X_ELE + 3 * QK_ELE) * sizeof(__bf16)) return;

  __bf16* wt = (__bf16*)d_ws;
  __bf16* Xb = wt + WT_ELE;
  __bf16* Qb = Xb + X_ELE;
  __bf16* Kb = Qb + QK_ELE;
  __bf16* Vt = Kb + QK_ELE;

  xcvt_kernel<<<dim3((unsigned)(X_ELE / (256 * 8))), 256, 0, stream>>>(x, Xb);
  wt_transpose_kernel<<<dim3(DE_ / 32, DIM_ / 32, 6 * H_), dim3(32, 8), 0, stream>>>(
      Wq, Wk, Wv, Wqs, Wks, Wvs, wt);
  proj_kernel<<<dim3(T_ / 16, H_, 3), 256, 0, stream>>>(Xb, wt, Qb, Kb, Vt);
  attn_kernel<<<dim3((H_ * (T_ / 16)) / 4), 128, 0, stream>>>(
      Qb, Kb, Vt, lq1, lq2, lk1, lk2, gam, bet, (float*)d_out);
}
